// Attention_1760936591725
// MI455X (gfx1250) — matrix operand order repaired and checked by reading, not compiled, not run on hardware
//
#include <hip/hip_runtime.h>
#include <hip/hip_bf16.h>
#include <cstdint>

typedef __attribute__((ext_vector_type(16))) _Float16 v16h;
typedef __attribute__((ext_vector_type(8)))  _Float16 h8;
typedef __attribute__((ext_vector_type(8)))  float    v8f;

#define DMODEL 4096
#define NHEAD  32
#define NKVH   8
#define DHEAD  128
#define DKV    (NKVH * DHEAD)     // 1024
#define BATCH  2
#define SEQT   2048
#define ROWS   (BATCH * SEQT)     // 4096

// ---- fragment loaders ------------------------------------------------------
// A 16x32 f16 layout: lane<16 holds K = 0..7 & 16..23 ; lane>=16 holds 8..15 & 24..31
static __device__ __forceinline__ v16h load_frag_a(const _Float16* p) {
  v16h a;
  ((h8*)&a)[0] = *(const h8*)(p);
  ((h8*)&a)[1] = *(const h8*)(p + 16);
  return a;
}
// B 32x16 f16 layout: lane<16 holds K = 0..7 & 16..23 ; lane>=16 holds 8..15 & 24..31
static __device__ __forceinline__ v16h load_frag_b(const _Float16* p) {
  v16h b;
  ((h8*)&b)[0] = *(const h8*)(p);
  ((h8*)&b)[1] = *(const h8*)(p + 16);
  return b;
}
static __device__ __forceinline__ v8f wmma_f16(v16h a, v16h b, v8f c) {
  return __builtin_amdgcn_wmma_f32_16x16x32_f16(false, a, false, b, (short)0, c,
                                                false, false);
}

// ---- conversion kernels ----------------------------------------------------
__global__ void f32_to_f16_kernel(const float* __restrict__ x,
                                  _Float16* __restrict__ y, size_t n) {
  size_t i = (size_t)blockIdx.x * blockDim.x + threadIdx.x;
  size_t st = (size_t)gridDim.x * blockDim.x;
  for (; i < n; i += st) y[i] = (_Float16)x[i];
}

// w[K][N] (row-major) -> wt[N][K] (row-major), f32 -> f16
__global__ void trans_f32_to_f16_kernel(const float* __restrict__ w,
                                        _Float16* __restrict__ wt,
                                        int K, int N) {
  size_t n = (size_t)K * N;
  size_t i = (size_t)blockIdx.x * blockDim.x + threadIdx.x;
  size_t st = (size_t)gridDim.x * blockDim.x;
  for (; i < n; i += st) {
    size_t k = i / (size_t)N;
    size_t c = i % (size_t)N;
    wt[c * (size_t)K + k] = (_Float16)w[i];
  }
}

// ---- NT GEMM: C[M,N] = A[M,K] * Bt[N,K]^T  (both f16 row-major) ------------
template <int OUTF16>
__global__ __launch_bounds__(256)
void gemm_nt_kernel(const _Float16* __restrict__ A,
                    const _Float16* __restrict__ Bt,
                    float* __restrict__ Cf, _Float16* __restrict__ Ch,
                    int M, int N, int K) {
  const int lane = threadIdx.x & 31;
  const int wave = threadIdx.x >> 5;
  const int l16  = lane & 15;
  const int half = lane >> 4;

  const int bm = blockIdx.y * 128 + (wave & 1) * 64;   // 2 waves along M
  const int bn = blockIdx.x * 128 + (wave >> 1) * 32;  // 4 waves along N

  v8f acc[4][2];
#pragma unroll
  for (int i = 0; i < 4; ++i)
#pragma unroll
    for (int j = 0; j < 2; ++j) acc[i][j] = (v8f)(0.0f);

  for (int k0 = 0; k0 < K; k0 += 32) {
    v16h af[4], bf[2];
#pragma unroll
    for (int i = 0; i < 4; ++i) {
      const _Float16* pa =
          A + (size_t)(bm + i * 16 + l16) * K + k0 + half * 8;
      af[i] = load_frag_a(pa);
      if (k0 + 32 < K) __builtin_prefetch(pa + 32, 0, 1);
    }
#pragma unroll
    for (int j = 0; j < 2; ++j) {
      const _Float16* pb =
          Bt + (size_t)(bn + j * 16 + l16) * K + k0 + half * 8;
      bf[j] = load_frag_b(pb);
      if (k0 + 32 < K) __builtin_prefetch(pb + 32, 0, 1);
    }
#pragma unroll
    for (int i = 0; i < 4; ++i)
#pragma unroll
      for (int j = 0; j < 2; ++j) acc[i][j] = wmma_f16(af[i], bf[j], acc[i][j]);
  }

#pragma unroll
  for (int i = 0; i < 4; ++i)
#pragma unroll
    for (int j = 0; j < 2; ++j)
#pragma unroll
      for (int r = 0; r < 8; ++r) {
        int row = bm + i * 16 + r + 8 * half;
        int col = bn + j * 16 + l16;
        size_t idx = (size_t)row * N + col;
        if (OUTF16) Ch[idx] = (_Float16)acc[i][j][r];
        else        Cf[idx] = acc[i][j][r];
      }
}

// ---- RoPE (in place on f16 activations) ------------------------------------
__global__ void rope_kernel(_Float16* __restrict__ qk,
                            const float* __restrict__ cs,
                            const float* __restrict__ sn,
                            int nHeads, size_t count) {
  size_t i = (size_t)blockIdx.x * blockDim.x + threadIdx.x;
  size_t st = (size_t)gridDim.x * blockDim.x;
  const int rowStride = nHeads * DHEAD;
  for (; i < count; i += st) {
    int p   = (int)(i & 63);
    int h   = (int)((i >> 6) % (size_t)nHeads);
    size_t row = i / (size_t)(64 * nHeads);
    int t = (int)(row % SEQT);
    size_t base = row * (size_t)rowStride + (size_t)h * DHEAD + 2 * p;
    float c = cs[t * 64 + p];
    float s = sn[t * 64 + p];
    float xr = (float)qk[base];
    float xi = (float)qk[base + 1];
    qk[base]     = (_Float16)(xr * c - xi * s);
    qk[base + 1] = (_Float16)(xr * s + xi * c);
  }
}

// ---- Flash attention: one block = (b, h, 64 q-rows), 4 waves ---------------
__global__ __launch_bounds__(128)
void attn_kernel(const _Float16* __restrict__ Q,
                 const _Float16* __restrict__ Kb,
                 const _Float16* __restrict__ Vb,
                 _Float16* __restrict__ Z) {
  __shared__ _Float16 Ks[32 * 128];       // K tile, row-major [key][d]
  __shared__ _Float16 Vt[128 * 32];       // V tile transposed [d][key]
  __shared__ _Float16 Ps[4][16 * 32];     // per-wave P staging

  const int lane = threadIdx.x & 31;
  const int wave = threadIdx.x >> 5;
  const int l16  = lane & 15;
  const int half = lane >> 4;

  const int nqt = SEQT / 64;                       // 32 q-tiles
  const int qt = blockIdx.x % nqt;
  const int h  = (blockIdx.x / nqt) % NHEAD;
  const int b  = blockIdx.x / (nqt * NHEAD);
  const int q0 = qt * 64;
  const int hkv = h >> 2;                          // GQA: 4 q-heads / kv-head

  const _Float16* Qh = Q  + (size_t)b * SEQT * DMODEL + (size_t)h   * DHEAD;
  const _Float16* Kh = Kb + (size_t)b * SEQT * DKV    + (size_t)hkv * DHEAD;
  const _Float16* Vh = Vb + (size_t)b * SEQT * DKV    + (size_t)hkv * DHEAD;

  // Q fragments for this wave's 16 rows (kept in registers for whole loop)
  const int qrow = q0 + wave * 16 + l16;
  v16h qf[4];
#pragma unroll
  for (int kk = 0; kk < 4; ++kk)
    qf[kk] = load_frag_a(Qh + (size_t)qrow * DMODEL + kk * 32 + half * 8);

  v8f of[8];
#pragma unroll
  for (int j = 0; j < 8; ++j) of[j] = (v8f)(0.0f);
  float m[8], l[8];
#pragma unroll
  for (int r = 0; r < 8; ++r) { m[r] = -1e30f; l[r] = 0.0f; }

  const float scale = 0.08838834764831845f;        // 1/sqrt(128)
  const int nkb = q0 / 32 + 2;                     // causal: keys <= q0+63

  for (int kb = 0; kb < nkb; ++kb) {
    const int kbase = kb * 32;
    __syncthreads();
    // Stage K tile with CDNA5 async global->LDS DMA (byte-exact, no VGPR
    // bounce); V tile is transposed through registers into Vt concurrently.
#pragma unroll
    for (int it = 0; it < 4; ++it) {
      int ci  = threadIdx.x + it * 128;            // 0..511 chunks of 8 f16
      int key = ci >> 4;
      int off = (ci & 15) * 8;
      const _Float16* gk = Kh + (size_t)(kbase + key) * DKV + off;
      uint32_t lk = (uint32_t)(uintptr_t)(&Ks[key * 128 + off]);
      asm volatile("global_load_async_to_lds_b128 %0, %1, off"
                   :: "v"(lk), "v"(gk) : "memory");
      h8 vv = *(const h8*)(Vh + (size_t)(kbase + key) * DKV + off);
#pragma unroll
      for (int e = 0; e < 8; ++e) Vt[(off + e) * 32 + key] = vv[e];
    }
    asm volatile("s_wait_asynccnt 0" ::: "memory");  // async transfers landed
    __syncthreads();

    // S = Q * K^T for two 16-key tiles
    v8f s[2];
#pragma unroll
    for (int kt = 0; kt < 2; ++kt) {
      s[kt] = (v8f)(0.0f);
#pragma unroll
      for (int kk = 0; kk < 4; ++kk) {
        v16h bf = load_frag_b(&Ks[(kt * 16 + l16) * 128 + kk * 32 + half * 8]);
        s[kt] = wmma_f16(qf[kk], bf, s[kt]);
      }
    }
    // scale + causal mask (C layout: col = l16, row = r + 8*half)
#pragma unroll
    for (int kt = 0; kt < 2; ++kt)
#pragma unroll
      for (int r = 0; r < 8; ++r) {
        int col  = kbase + kt * 16 + l16;
        int rowa = q0 + wave * 16 + r + 8 * half;
        float v = s[kt][r] * scale;
        if (col > rowa) v = -1e30f;
        s[kt][r] = v;
      }
    // online softmax, per-row reductions across the 16-lane half-group
    float p0[8], p1[8];
#pragma unroll
    for (int r = 0; r < 8; ++r) {
      float mx = fmaxf(s[0][r], s[1][r]);
#pragma unroll
      for (int off = 8; off >= 1; off >>= 1)
        mx = fmaxf(mx, __shfl_xor(mx, off, 32));
      float mn = fmaxf(m[r], mx);
      p0[r] = __expf(s[0][r] - mn);
      p1[r] = __expf(s[1][r] - mn);
      float rs = p0[r] + p1[r];
#pragma unroll
      for (int off = 8; off >= 1; off >>= 1) rs += __shfl_xor(rs, off, 32);
      float alpha = __expf(m[r] - mn);
      l[r] = l[r] * alpha + rs;
      m[r] = mn;
#pragma unroll
      for (int j = 0; j < 8; ++j) of[j][r] *= alpha;
    }
    // stage P (C layout -> LDS), then reload as A fragment
#pragma unroll
    for (int r = 0; r < 8; ++r) {
      int prow = r + 8 * half;
      Ps[wave][prow * 32 + l16]      = (_Float16)p0[r];
      Ps[wave][prow * 32 + 16 + l16] = (_Float16)p1[r];
    }
    asm volatile("s_wait_dscnt 0" ::: "memory");   // same-wave LDS RAW
    v16h pf = load_frag_a(&Ps[wave][l16 * 32 + half * 8]);
    // O += P * V
#pragma unroll
    for (int j = 0; j < 8; ++j) {
      v16h bf = load_frag_b(&Vt[(j * 16 + l16) * 32 + half * 8]);
      of[j] = wmma_f16(pf, bf, of[j]);
    }
  }

  // epilogue: normalize and store z[b*T + row][h*128 + d] as f16
#pragma unroll
  for (int j = 0; j < 8; ++j)
#pragma unroll
    for (int r = 0; r < 8; ++r) {
      int row = q0 + wave * 16 + r + 8 * half;
      int col = h * DHEAD + j * 16 + l16;
      Z[(size_t)(b * SEQT + row) * DMODEL + col] = (_Float16)(of[j][r] / l[r]);
    }
}

// ---- host-side orchestration ----------------------------------------------
extern "C" void kernel_launch(void* const* d_in, const int* in_sizes, int n_in,
                              void* d_out, int out_size, void* d_ws,
                              size_t ws_size, hipStream_t stream) {
  const float* x    = (const float*)d_in[0];
  const float* fcos = (const float*)d_in[1];
  const float* fsin = (const float*)d_in[2];
  const float* wq   = (const float*)d_in[3];
  const float* wk   = (const float*)d_in[4];
  const float* wv   = (const float*)d_in[5];
  const float* wo   = (const float*)d_in[6];
  float* out = (float*)d_out;

  char* w = (char*)d_ws;
  _Float16* xh  = (_Float16*)w; w += (size_t)ROWS * DMODEL * 2;
  _Float16* wqT = (_Float16*)w; w += (size_t)DMODEL * DMODEL * 2;
  _Float16* wkT = (_Float16*)w; w += (size_t)DKV * DMODEL * 2;
  _Float16* wvT = (_Float16*)w; w += (size_t)DKV * DMODEL * 2;
  _Float16* woT = (_Float16*)w; w += (size_t)DMODEL * DMODEL * 2;
  _Float16* Qb  = (_Float16*)w; w += (size_t)ROWS * DMODEL * 2;
  _Float16* Kb  = (_Float16*)w; w += (size_t)ROWS * DKV * 2;
  _Float16* Vb  = (_Float16*)w; w += (size_t)ROWS * DKV * 2;
  _Float16* Zb  = (_Float16*)w; w += (size_t)ROWS * DMODEL * 2;

  // 1) precision conversion + weight transposes
  f32_to_f16_kernel<<<4096, 256, 0, stream>>>(x, xh, (size_t)ROWS * DMODEL);
  trans_f32_to_f16_kernel<<<4096, 256, 0, stream>>>(wq, wqT, DMODEL, DMODEL);
  trans_f32_to_f16_kernel<<<2048, 256, 0, stream>>>(wk, wkT, DMODEL, DKV);
  trans_f32_to_f16_kernel<<<2048, 256, 0, stream>>>(wv, wvT, DMODEL, DKV);
  trans_f32_to_f16_kernel<<<4096, 256, 0, stream>>>(wo, woT, DMODEL, DMODEL);

  // 2) QKV projections (WMMA GEMM, f16 out)
  gemm_nt_kernel<1><<<dim3(DMODEL / 128, ROWS / 128), 256, 0, stream>>>(
      xh, wqT, nullptr, Qb, ROWS, DMODEL, DMODEL);
  gemm_nt_kernel<1><<<dim3(DKV / 128, ROWS / 128), 256, 0, stream>>>(
      xh, wkT, nullptr, Kb, ROWS, DKV, DMODEL);
  gemm_nt_kernel<1><<<dim3(DKV / 128, ROWS / 128), 256, 0, stream>>>(
      xh, wvT, nullptr, Vb, ROWS, DKV, DMODEL);

  // 3) RoPE on Q and K
  rope_kernel<<<4096, 256, 0, stream>>>(Qb, fcos, fsin, NHEAD,
                                        (size_t)ROWS * NHEAD * 64);
  rope_kernel<<<1024, 256, 0, stream>>>(Kb, fcos, fsin, NKVH,
                                        (size_t)ROWS * NKVH * 64);

  // 4) causal flash attention (WMMA + async K staging)
  attn_kernel<<<BATCH * NHEAD * (SEQT / 64), 128, 0, stream>>>(Qb, Kb, Vb, Zb);

  // 5) output projection (WMMA GEMM, f32 out)
  gemm_nt_kernel<0><<<dim3(DMODEL / 128, ROWS / 128), 256, 0, stream>>>(
      Zb, woT, out, nullptr, ROWS, DMODEL, DMODEL);
}
